// SLOG_N_56014963474694
// MI455X (gfx1250) — matrix operand order repaired and checked by reading, not compiled, not run on hardware
//
#include <hip/hip_runtime.h>
#include <hip/hip_bf16.h>

typedef __attribute__((ext_vector_type(16))) __bf16 v16bf;
typedef __attribute__((ext_vector_type(8)))  __bf16 v8bf;
typedef __attribute__((ext_vector_type(4)))  __bf16 v4bf;
typedef __attribute__((ext_vector_type(8)))  float  v8f;

#define GTILE_M 64
#define GTILE_K 32
#define GLDA 40   // padded LDS row stride (bf16 elems): 80B, 16B-aligned slices
#define GLDB 40

__device__ __forceinline__ v8bf bf8_zero() {
  v8bf z;
#pragma unroll
  for (int i = 0; i < 8; ++i) z[i] = (__bf16)0.0f;
  return z;
}

__device__ __forceinline__ v8bf cvt8(const float* p) {
  const float4* p4 = (const float4*)p;
  const float4 f0 = p4[0], f1 = p4[1];
  v8bf v;
  v[0] = (__bf16)f0.x; v[1] = (__bf16)f0.y; v[2] = (__bf16)f0.z; v[3] = (__bf16)f0.w;
  v[4] = (__bf16)f1.x; v[5] = (__bf16)f1.y; v[6] = (__bf16)f1.z; v[7] = (__bf16)f1.w;
  return v;
}

// Load one 64x32 A tile (of op(U)) + 32x64 B tile into registers.
// B panel is stored COLUMN-MAJOR in global (Bt[d][n]) so both its global
// loads and LDS stores are contiguous 128-bit ops.
// GUARD=false is the fast path for interior tiles (no bounds checks).
template<bool TRANS, bool PACKED, bool GUARD>
__device__ __forceinline__ void load_tile_regs(
    const float* __restrict__ Uf, const __bf16* __restrict__ Ub,
    const __bf16* __restrict__ Bt, int n, int d, int row0, int col0, int k0,
    int t, v8bf areg[2], v8bf breg[2])
{
#pragma unroll
  for (int p = 0; p < 2; ++p) {
    const int slot = t + p * 128;
    v8bf v = bf8_zero();
    if (!TRANS) {
      const int mi = slot >> 2, ki = (slot & 3) << 3;
      const int gm = row0 + mi, gk = k0 + ki;
      if (!GUARD || (gm < n && gk + 8 <= n)) {   // n % 8 == 0 -> exact
        if (PACKED) v = *(const v8bf*)(Ub + (size_t)gm * n + gk);
        else        v = cvt8(Uf + (size_t)gm * n + gk);
      }
    } else {
      const int ki = slot >> 3, mi0 = (slot & 7) << 3;
      const int gk = k0 + ki, gm0 = row0 + mi0;
      if (!GUARD || (gk < n && gm0 + 8 <= n)) {
        if (PACKED) v = *(const v8bf*)(Ub + (size_t)gk * n + gm0);
        else        v = cvt8(Uf + (size_t)gk * n + gm0);
      }
    }
    areg[p] = v;
  }
#pragma unroll
  for (int p = 0; p < 2; ++p) {
    const int slot = t + p * 128;
    const int col = slot >> 2;             // 0..63
    const int kr  = (slot & 3) << 3;       // 0,8,16,24
    const int gk = k0 + kr;
    v8bf v = bf8_zero();
    if (!GUARD || gk + 8 <= n)             // n % 8 == 0 -> exact
      v = *(const v8bf*)(Bt + (size_t)(col0 + col) * n + gk);
    breg[p] = v;
  }
}

template<bool TRANS>
__device__ __forceinline__ void store_tile_lds(
    __bf16* ldsA, __bf16* ldsB, int t, const v8bf areg[2], const v8bf breg[2])
{
#pragma unroll
  for (int p = 0; p < 2; ++p) {
    const int slot = t + p * 128;
    if (!TRANS) {
      const int mi = slot >> 2, ki = (slot & 3) << 3;
      *(v8bf*)(ldsA + mi * GLDA + ki) = areg[p];
    } else {
      const int ki = slot >> 3, mi0 = (slot & 7) << 3;
#pragma unroll
      for (int j = 0; j < 8; ++j) ldsA[(mi0 + j) * GLDA + ki] = areg[p][j];
    }
  }
#pragma unroll
  for (int p = 0; p < 2; ++p) {
    const int slot = t + p * 128;
    const int col = slot >> 2, kr = (slot & 3) << 3;
    *(v8bf*)(ldsB + col * GLDB + kr) = breg[p];
  }
}

__device__ __forceinline__ void compute_tile(
    const __bf16* ldsA, const __bf16* ldsB, int wave, int ml, int half,
    v8f (&acc)[4])
{
  // A fragment (16x32 bf16): lane<16 -> K 0..15, lane>=16 -> K 16..31
  const int m  = wave * 16 + ml;
  const int kb = half * 16;
  const v8bf alo = *(const v8bf*)(ldsA + m * GLDA + kb);
  const v8bf ahi = *(const v8bf*)(ldsA + m * GLDA + kb + 8);
  const v16bf afrag = __builtin_shufflevector(alo, ahi,
      0,1,2,3,4,5,6,7,8,9,10,11,12,13,14,15);
#pragma unroll
  for (int ct = 0; ct < 4; ++ct) {
    // B fragment (32x16 bf16): lane<16 -> K 0..15, lane>=16 -> K 16..31, col = lane%16
    const int c   = ct * 16 + ml;
    const int kb2 = half * 16;
    const v8bf blo = *(const v8bf*)(ldsB + c * GLDB + kb2);
    const v8bf bhi = *(const v8bf*)(ldsB + c * GLDB + kb2 + 8);
    const v16bf bfrag = __builtin_shufflevector(blo, bhi,
        0,1,2,3,4,5,6,7,8,9,10,11,12,13,14,15);
    acc[ct] = __builtin_amdgcn_wmma_f32_16x16x32_bf16(
        false, afrag, false, bfrag, (short)0, acc[ct], false, false);
  }
}

// C[n x d] = op(U) @ B  (op = U or U^T), B bf16 column-major [d x n], C fp32.
// 128 threads = 4 wave32s; each wave owns 16 rows x 64 cols.
// Double-buffered LDS: global loads for tile s issue before the WMMAs of
// tile s-1; one barrier per k-step.
template<bool TRANS, bool PACKED>
__global__ void __launch_bounds__(128)
gemm_spectral(const float* __restrict__ Uf, const __bf16* __restrict__ Ub,
              const __bf16* __restrict__ Bt, float* __restrict__ C,
              int n, int d)
{
  __shared__ __align__(16) __bf16 ldsA[2][GTILE_M * GLDA];
  __shared__ __align__(16) __bf16 ldsB[2][64 * GLDB];

  const int t    = threadIdx.x;
  const int lane = t & 31;
  const int wave = t >> 5;
  const int row0 = blockIdx.x * GTILE_M;
  const int col0 = blockIdx.y * 64;
  const int half = (lane < 16) ? 0 : 1;
  const int ml   = lane & 15;
  const bool fullRows = (row0 + GTILE_M <= n);
  const int T = (n + GTILE_K - 1) / GTILE_K;

  v8f acc[4];
#pragma unroll
  for (int ct = 0; ct < 4; ++ct)
#pragma unroll
    for (int r = 0; r < 8; ++r) acc[ct][r] = 0.0f;

  v8bf areg[2], breg[2];

  if (fullRows)
    load_tile_regs<TRANS, PACKED, false>(Uf, Ub, Bt, n, d, row0, col0, 0, t, areg, breg);
  else
    load_tile_regs<TRANS, PACKED, true >(Uf, Ub, Bt, n, d, row0, col0, 0, t, areg, breg);
  store_tile_lds<TRANS>(&ldsA[0][0], &ldsB[0][0], t, areg, breg);
  __syncthreads();

  int buf = 0;
  for (int s = 1; s < T; ++s) {
    const int k0 = s * GTILE_K;
    // prefetch the U stream two tiles ahead (dominant HBM traffic)
    const int kpf = k0 + 2 * GTILE_K;
    if (kpf + GTILE_K <= n) {
      if (!TRANS) {
        const int mi = t >> 2, ki = (t & 3) << 3;
        if (PACKED) __builtin_prefetch(Ub + (size_t)(row0 + mi) * n + kpf + ki, 0, 1);
        else        __builtin_prefetch(Uf + (size_t)(row0 + mi) * n + kpf + ki, 0, 1);
      } else {
        const int ki = t >> 3, mi0 = (t & 7) << 3;
        if (PACKED) __builtin_prefetch(Ub + (size_t)(kpf + ki) * n + row0 + mi0, 0, 1);
        else        __builtin_prefetch(Uf + (size_t)(kpf + ki) * n + row0 + mi0, 0, 1);
      }
    }
    if (fullRows && k0 + GTILE_K <= n)
      load_tile_regs<TRANS, PACKED, false>(Uf, Ub, Bt, n, d, row0, col0, k0, t, areg, breg);
    else
      load_tile_regs<TRANS, PACKED, true >(Uf, Ub, Bt, n, d, row0, col0, k0, t, areg, breg);
    compute_tile(&ldsA[buf][0], &ldsB[buf][0], wave, ml, half, acc);
    store_tile_lds<TRANS>(&ldsA[buf ^ 1][0], &ldsB[buf ^ 1][0], t, areg, breg);
    __syncthreads();
    buf ^= 1;
  }
  compute_tile(&ldsA[buf][0], &ldsB[buf][0], wave, ml, half, acc);

  // C/D layout: VGPR r = row (r + 8*half), col = lane%16
  const int rowBase = row0 + wave * 16 + half * 8;
#pragma unroll
  for (int ct = 0; ct < 4; ++ct) {
    const int c = col0 + ct * 16 + ml;
#pragma unroll
    for (int r = 0; r < 8; ++r) {
      const int row = rowBase + r;
      if (row < n) C[(size_t)row * d + c] = acc[ct][r];
    }
  }
}

// v1 = La^va ; v2 = ((2*(La-1e-8)-1)^2 + 1)^vb
__global__ void pow_kernel(const float* __restrict__ La, const float* __restrict__ va,
                           const float* __restrict__ vb, float* __restrict__ v1,
                           float* __restrict__ v2, int n) {
  int i = blockIdx.x * blockDim.x + threadIdx.x;
  if (i >= n) return;
  float a = *va, b = *vb, l = La[i];
  v1[i] = powf(l, a);
  float u = 2.0f * (l - 1e-8f) - 1.0f;
  v2[i] = powf(u * u + 1.0f, b);
}

// Bt[d][n] (column-major bf16 panel) = scale[row] * src[n][d]  (scale may be null)
// One thread per (column, 8-row run): coalesced fp32 reads, one b128 store.
__global__ void cvt_scale_t(const float* __restrict__ src, const float* __restrict__ scale,
                            __bf16* __restrict__ dst, int n, int d) {
  int idx = blockIdx.x * blockDim.x + threadIdx.x;
  const int total = (n >> 3) * d;
  if (idx >= total) return;
  const int c  = idx % d;
  const int r0 = (idx / d) << 3;
  v8bf v;
#pragma unroll
  for (int j = 0; j < 8; ++j) {
    const float s = scale ? scale[r0 + j] : 1.0f;
    v[j] = (__bf16)(s * src[(size_t)(r0 + j) * d + c]);
  }
  *(v8bf*)(dst + (size_t)c * n + r0) = v;
}

__global__ void cvt_u(const float* __restrict__ U, __bf16* __restrict__ Ub, long long total) {
  long long i = ((long long)blockIdx.x * blockDim.x + threadIdx.x) * 4;
  if (i + 4 > total) return;
  const float4 f = *(const float4*)(U + i);
  v4bf o;
  o[0] = (__bf16)f.x; o[1] = (__bf16)f.y; o[2] = (__bf16)f.z; o[3] = (__bf16)f.w;
  *(v4bf*)(Ub + i) = o;
}

// Y[n x dout] = X[n x din] @ W[din x dout] + bias
__global__ void linear_k(const float* __restrict__ X, const float* __restrict__ W,
                         const float* __restrict__ bias, float* __restrict__ Y,
                         int n, int din, int dout) {
  int idx = blockIdx.x * blockDim.x + threadIdx.x;
  if (idx >= n * dout) return;
  int i = idx / dout, o = idx % dout;
  float s = bias[o];
  const float* xr = X + (size_t)i * din;
  for (int k = 0; k < din; ++k) s += xr[k] * W[(size_t)k * dout + o];
  Y[idx] = s;
}

// per-column mean + rsqrt(var+eps), biased variance; one block per column
__global__ void bn_stats(const float* __restrict__ H, float* __restrict__ mu,
                         float* __restrict__ rs, int n, int d) {
  __shared__ float ssum[256], ssq[256];
  const int c = blockIdx.x, tid = threadIdx.x;
  float s = 0.f, q = 0.f;
  for (int i = tid; i < n; i += blockDim.x) {
    float v = H[(size_t)i * d + c];
    s += v; q += v * v;
  }
  ssum[tid] = s; ssq[tid] = q;
  __syncthreads();
  for (int w = 128; w > 0; w >>= 1) {
    if (tid < w) { ssum[tid] += ssum[tid + w]; ssq[tid] += ssq[tid + w]; }
    __syncthreads();
  }
  if (tid == 0) {
    float m = ssum[0] / n;
    float var = ssq[0] / n - m * m;
    mu[c] = m;
    rs[c] = rsqrtf(var + 1e-5f);
  }
}

// Xn = relu( relu(g*(H-mu)*rs + b) + L )
__global__ void bn1_fuse(const float* __restrict__ H, const float* __restrict__ mu,
                         const float* __restrict__ rs, const float* __restrict__ g,
                         const float* __restrict__ b, const float* __restrict__ L,
                         float* __restrict__ Xn, int n, int d) {
  int i = blockIdx.x * blockDim.x + threadIdx.x;
  if (i >= n * d) return;
  int c = i % d;
  float h = fmaxf(g[c] * (H[i] - mu[c]) * rs[c] + b[c], 0.0f);
  Xn[i] = fmaxf(h + L[i], 0.0f);
}

// Y = g*(X-mu)*rs + b   (optionally relu)
__global__ void bn_apply(const float* __restrict__ X, const float* __restrict__ mu,
                         const float* __restrict__ rs, const float* __restrict__ g,
                         const float* __restrict__ b, float* __restrict__ Y,
                         int n, int d, int do_relu) {
  int i = blockIdx.x * blockDim.x + threadIdx.x;
  if (i >= n * d) return;
  int c = i % d;
  float v = g[c] * (X[i] - mu[c]) * rs[c] + b[c];
  Y[i] = do_relu ? fmaxf(v, 0.0f) : v;
}

// out = log_softmax(sigmoid(O1) + O2), rows of width d (<=64)
__global__ void out_kernel(const float* __restrict__ O1, const float* __restrict__ O2,
                           float* __restrict__ out, int n, int d) {
  int i = blockIdx.x * blockDim.x + threadIdx.x;
  if (i >= n) return;
  float tmp[64];
  float m = -3.4e38f;
  for (int c = 0; c < d; ++c) {
    float v = 1.0f / (1.0f + expf(-O1[(size_t)i * d + c])) + O2[(size_t)i * d + c];
    tmp[c] = v;
    m = fmaxf(m, v);
  }
  float s = 0.f;
  for (int c = 0; c < d; ++c) s += expf(tmp[c] - m);
  float lse = m + logf(s);
  for (int c = 0; c < d; ++c) out[(size_t)i * d + c] = tmp[c] - lse;
}

extern "C" void kernel_launch(void* const* d_in, const int* in_sizes, int n_in,
                              void* d_out, int out_size, void* d_ws, size_t ws_size,
                              hipStream_t stream) {
  const float* X  = (const float*)d_in[0];
  const float* La = (const float*)d_in[1];
  const float* U  = (const float*)d_in[2];
  const int n    = in_sizes[1];            // 10000
  const int cin  = in_sizes[0] / n;        // 128
  const int hid  = in_sizes[6];            // 64
  const int cout = in_sizes[30];           // 40

  const float* va[3]  = {(const float*)d_in[3],  (const float*)d_in[13], (const float*)d_in[23]};
  const float* vb[3]  = {(const float*)d_in[4],  (const float*)d_in[14], (const float*)d_in[24]};
  const float* Wbw[3] = {(const float*)d_in[5],  (const float*)d_in[15], (const float*)d_in[25]};
  const float* Wbb[3] = {(const float*)d_in[6],  (const float*)d_in[16], (const float*)d_in[26]};
  const float* g1[3]  = {(const float*)d_in[7],  (const float*)d_in[17], (const float*)d_in[27]};
  const float* b1[3]  = {(const float*)d_in[8],  (const float*)d_in[18], (const float*)d_in[28]};
  const float* lw[3]  = {(const float*)d_in[9],  (const float*)d_in[19], (const float*)d_in[29]};
  const float* lb[3]  = {(const float*)d_in[10], (const float*)d_in[20], (const float*)d_in[30]};
  const float* g2[2]  = {(const float*)d_in[11], (const float*)d_in[21]};
  const float* b2[2]  = {(const float*)d_in[12], (const float*)d_in[22]};
  const float* mlpw = (const float*)d_in[31];
  const float* mlpb = (const float*)d_in[32];

  // ---- workspace layout ----
  char* base = (char*)d_ws;
  size_t off = 0;
  auto alloc = [&](size_t bytes) -> void* {
    off = (off + 255) & ~(size_t)255;
    void* p = base + off;
    off += bytes;
    return p;
  };
  float*  v1  = (float*)alloc((size_t)n * 4);
  float*  v2  = (float*)alloc((size_t)n * 4);
  float*  tA  = (float*)alloc((size_t)n * cin * 4);
  float*  tB  = (float*)alloc((size_t)n * cin * 4);
  __bf16* Bbf = (__bf16*)alloc((size_t)n * cin * 2);
  float*  H   = (float*)alloc((size_t)n * hid * 4);
  float*  Lb  = (float*)alloc((size_t)n * hid * 4);
  float*  Xn  = (float*)alloc((size_t)n * hid * 4);
  float*  X1  = (float*)alloc((size_t)n * hid * 4);
  float*  X2  = (float*)alloc((size_t)n * hid * 4);
  float*  O1  = (float*)alloc((size_t)n * cout * 4);
  float*  O2  = (float*)alloc((size_t)n * cout * 4);
  float*  mu  = (float*)alloc(256);
  float*  rs  = (float*)alloc(256);
  float*  mu2 = (float*)alloc(256);
  float*  rs2 = (float*)alloc(256);

  const size_t ub_bytes = (size_t)n * n * 2;
  size_t ub_off = (off + 255) & ~(size_t)255;
  const bool packed = (ub_off + ub_bytes) <= ws_size;
  __bf16* Ubf = packed ? (__bf16*)(base + ub_off) : nullptr;

  auto nb = [](long long total, int tpb) { return (unsigned)((total + tpb - 1) / tpb); };

  if (packed)
    cvt_u<<<nb((long long)n * n / 4, 256), 256, 0, stream>>>(U, Ubf, (long long)n * n);

  auto gemm = [&](bool trans, const __bf16* Bt, float* C, int d) {
    dim3 g((n + GTILE_M - 1) / GTILE_M, d / 64);
    if (trans) {
      if (packed) gemm_spectral<true,  true ><<<g, 128, 0, stream>>>(U, Ubf, Bt, C, n, d);
      else        gemm_spectral<true,  false><<<g, 128, 0, stream>>>(U, Ubf, Bt, C, n, d);
    } else {
      if (packed) gemm_spectral<false, true ><<<g, 128, 0, stream>>>(U, Ubf, Bt, C, n, d);
      else        gemm_spectral<false, false><<<g, 128, 0, stream>>>(U, Ubf, Bt, C, n, d);
    }
  };

  // spectral(Xin) = U @ (v2 . (U^T @ (U @ (v1 . (U^T @ Xin)))))  -> result in tB
  auto spectral = [&](const float* Xin, int d) {
    const unsigned cvtb = nb(((long long)n >> 3) * d, 256);
    cvt_scale_t<<<cvtb, 256, 0, stream>>>(Xin, nullptr, Bbf, n, d);
    gemm(true,  Bbf, tA, d);
    cvt_scale_t<<<cvtb, 256, 0, stream>>>(tA, v1, Bbf, n, d);
    gemm(false, Bbf, tB, d);
    cvt_scale_t<<<cvtb, 256, 0, stream>>>(tB, nullptr, Bbf, n, d);
    gemm(true,  Bbf, tA, d);
    cvt_scale_t<<<cvtb, 256, 0, stream>>>(tA, v2, Bbf, n, d);
    gemm(false, Bbf, tB, d);
  };

  const float* Xcur = X;
  int din = cin;
  float* Xbufs[2] = {X1, X2};
  for (int i = 0; i < 2; ++i) {
    pow_kernel<<<nb(n, 256), 256, 0, stream>>>(La, va[i], vb[i], v1, v2, n);
    spectral(Xcur, din);
    linear_k<<<nb((long long)n * hid, 256), 256, 0, stream>>>(tB, Wbw[i], Wbb[i], H, n, din, hid);
    linear_k<<<nb((long long)n * hid, 256), 256, 0, stream>>>(Xcur, lw[i], lb[i], Lb, n, din, hid);
    bn_stats<<<hid, 256, 0, stream>>>(H, mu, rs, n, hid);
    bn1_fuse<<<nb((long long)n * hid, 256), 256, 0, stream>>>(H, mu, rs, g1[i], b1[i], Lb, Xn, n, hid);
    bn_stats<<<hid, 256, 0, stream>>>(Xn, mu2, rs2, n, hid);
    bn_apply<<<nb((long long)n * hid, 256), 256, 0, stream>>>(Xn, mu2, rs2, g2[i], b2[i], Xbufs[i], n, hid, 0);
    Xcur = Xbufs[i];
    din = hid;
  }

  // final block
  pow_kernel<<<nb(n, 256), 256, 0, stream>>>(La, va[2], vb[2], v1, v2, n);
  spectral(Xcur, hid);
  linear_k<<<nb((long long)n * hid, 256), 256, 0, stream>>>(tB, Wbw[2], Wbb[2], H, n, hid, hid);
  bn_stats<<<hid, 256, 0, stream>>>(H, mu, rs, n, hid);
  bn_apply<<<nb((long long)n * hid, 256), 256, 0, stream>>>(H, mu, rs, g1[2], b1[2], Xn, n, hid, 1);
  linear_k<<<nb((long long)n * cout, 256), 256, 0, stream>>>(Xn, mlpw, mlpb, O1, n, hid, cout);
  linear_k<<<nb((long long)n * cout, 256), 256, 0, stream>>>(Xcur, lw[2], lb[2], O2, n, hid, cout);
  out_kernel<<<nb(n, 256), 256, 0, stream>>>(O1, O2, (float*)d_out, n, cout);
}
